// AttentionLayer_77043123356287
// MI455X (gfx1250) — matrix operand order repaired and checked by reading, not compiled, not run on hardware
//
#include <hip/hip_runtime.h>

typedef __attribute__((ext_vector_type(16))) _Float16 v16h;
typedef __attribute__((ext_vector_type(8)))  float    v8f;

#define F_FIELDS 8
#define WPB 4            // waves per block -> 96KB LDS, 3 blocks/WGP
#define LEAKY 0.01f

__device__ __forceinline__ float lrelu(float v) { return v >= 0.0f ? v : LEAKY * v; }

__global__ void __launch_bounds__(32 * WPB)
autoint_attn_kernel(const float* __restrict__ x,
                    const float* __restrict__ wq,
                    const float* __restrict__ wk,
                    const float* __restrict__ wv,
                    float* __restrict__ out,
                    int N, int ntiles)
{
    // per-wave Q/K/V scratch: [wave][proj][field][n][e], fp32
    __shared__ float qkv[WPB][3][F_FIELDS][16][16];

    const int tid  = threadIdx.x;
    const int wave = tid >> 5;
    const int lane = tid & 31;
    const int e    = lane & 15;   // B column / C column / A row
    const int h    = lane >> 4;   // lane half
    const int tile = blockIdx.x * WPB + wave;
    const bool active = tile < ntiles;
    const int n0 = tile * 16;

    if (active) {
        // ---------- Build B operands: W (64x16, K=d, N=e) in WMMA B layout.
        // half 0 lanes: K = c*32 + {0..7,16..23} ; half 1 lanes: K = c*32 + {8..15,24..31}
        // split f32 -> f16 hi + f16 lo for ~fp32-accurate WMMA.
        const float* wmat[3] = { wq, wk, wv };
        v16h bhi[3][2], blo[3][2];
        #pragma unroll
        for (int p = 0; p < 3; ++p) {
            #pragma unroll
            for (int c = 0; c < 2; ++c) {
                const int kbase = c * 32 + h * 8;
                v16h hi, lo;
                #pragma unroll
                for (int t = 0; t < 16; ++t) {
                    float w = wmat[p][(kbase + (t < 8 ? t : t + 8)) * 16 + e];
                    _Float16 wh = (_Float16)w;
                    hi[t] = wh;
                    lo[t] = (_Float16)(w - (float)wh);   // exact residual
                }
                bhi[p][c] = hi; blo[p][c] = lo;
            }
        }

        // ---------- Phase 1: Q/K/V projections via v_wmma_f32_16x16x32_f16
        const int row = (n0 + e < N) ? (n0 + e) : (N - 1);   // clamp partial tile
        #pragma unroll 1
        for (int f = 0; f < F_FIELDS; ++f) {
            const float* xr = x + ((size_t)f * N + row) * 64;
            if (f + 1 < F_FIELDS) {                 // warm next field's row
                __builtin_prefetch(xr + (size_t)N * 64, 0, 1);
                __builtin_prefetch(xr + (size_t)N * 64 + 32, 0, 1);
            }
            // A operand (16x32 f16): half0 K={0..7,16..23}+32c, half1 K={8..15,24..31}+32c
            v16h ahi[2], alo[2];
            #pragma unroll
            for (int c = 0; c < 2; ++c) {
                const int k0 = c * 32 + h * 8;
                float4 f0 = *(const float4*)(xr + k0);
                float4 f1 = *(const float4*)(xr + k0 + 4);
                float4 f2 = *(const float4*)(xr + k0 + 16);
                float4 f3 = *(const float4*)(xr + k0 + 20);
                float vals[16] = { f0.x,f0.y,f0.z,f0.w, f1.x,f1.y,f1.z,f1.w,
                                   f2.x,f2.y,f2.z,f2.w, f3.x,f3.y,f3.z,f3.w };
                #pragma unroll
                for (int t = 0; t < 16; ++t) {
                    _Float16 vh = (_Float16)vals[t];
                    ahi[c][t] = vh;
                    alo[c][t] = (_Float16)(vals[t] - (float)vh);
                }
            }
            #pragma unroll
            for (int p = 0; p < 3; ++p) {
                v8f acc = {};
                acc = __builtin_amdgcn_wmma_f32_16x16x32_f16(false, ahi[0], false, bhi[p][0], (short)0, acc, false, false);
                acc = __builtin_amdgcn_wmma_f32_16x16x32_f16(false, ahi[1], false, bhi[p][1], (short)0, acc, false, false);
                acc = __builtin_amdgcn_wmma_f32_16x16x32_f16(false, ahi[0], false, blo[p][0], (short)0, acc, false, false);
                acc = __builtin_amdgcn_wmma_f32_16x16x32_f16(false, ahi[1], false, blo[p][1], (short)0, acc, false, false);
                acc = __builtin_amdgcn_wmma_f32_16x16x32_f16(false, alo[0], false, bhi[p][0], (short)0, acc, false, false);
                acc = __builtin_amdgcn_wmma_f32_16x16x32_f16(false, alo[1], false, bhi[p][1], (short)0, acc, false, false);
                // C/D layout: VGPR j -> M = j + 8*h, N = e
                #pragma unroll
                for (int j = 0; j < 8; ++j)
                    qkv[wave][p][f][h * 8 + j][e] = acc[j];
            }
        }
    }

    __syncthreads();

    // ---------- Phase 2: masked softmax attention per (n,f); 4 pairs per lane
    if (active) {
        const float DIAG = 1.0f + (-1.0e12f);    // multiplicative diagonal mask
        #pragma unroll 1
        for (int i = 0; i < 4; ++i) {
            const int pr = lane + 32 * i;        // 0..127
            const int n  = pr >> 3;
            const int f  = pr & 7;

            const float4* qr = (const float4*)&qkv[wave][0][f][n][0];
            float4 q0 = qr[0], q1 = qr[1], q2 = qr[2], q3 = qr[3];

            float logits[8];
            #pragma unroll
            for (int g = 0; g < 8; ++g) {
                const float4* kr = (const float4*)&qkv[wave][1][g][n][0];
                float4 k0 = kr[0], k1 = kr[1], k2 = kr[2], k3 = kr[3];
                float s = q0.x*k0.x + q0.y*k0.y + q0.z*k0.z + q0.w*k0.w
                        + q1.x*k1.x + q1.y*k1.y + q1.z*k1.z + q1.w*k1.w
                        + q2.x*k2.x + q2.y*k2.y + q2.z*k2.z + q2.w*k2.w
                        + q3.x*k3.x + q3.y*k3.y + q3.z*k3.z + q3.w*k3.w;
                s *= (1.0f / 16.0f);
                logits[g] = (g == f) ? s * DIAG : s;
            }
            float m = logits[0];
            #pragma unroll
            for (int g = 1; g < 8; ++g) m = fmaxf(m, logits[g]);
            float ex[8], sum = 0.0f;
            #pragma unroll
            for (int g = 0; g < 8; ++g) { ex[g] = __expf(logits[g] - m); sum += ex[g]; }
            const float inv = 1.0f / sum;

            float4 o0 = {0,0,0,0}, o1 = {0,0,0,0}, o2 = {0,0,0,0}, o3 = {0,0,0,0};
            #pragma unroll
            for (int g = 0; g < 8; ++g) {
                const float c = ex[g] * inv;
                const float4* vr = (const float4*)&qkv[wave][2][g][n][0];
                float4 v0 = vr[0], v1 = vr[1], v2 = vr[2], v3 = vr[3];
                o0.x = fmaf(c, v0.x, o0.x); o0.y = fmaf(c, v0.y, o0.y);
                o0.z = fmaf(c, v0.z, o0.z); o0.w = fmaf(c, v0.w, o0.w);
                o1.x = fmaf(c, v1.x, o1.x); o1.y = fmaf(c, v1.y, o1.y);
                o1.z = fmaf(c, v1.z, o1.z); o1.w = fmaf(c, v1.w, o1.w);
                o2.x = fmaf(c, v2.x, o2.x); o2.y = fmaf(c, v2.y, o2.y);
                o2.z = fmaf(c, v2.z, o2.z); o2.w = fmaf(c, v2.w, o2.w);
                o3.x = fmaf(c, v3.x, o3.x); o3.y = fmaf(c, v3.y, o3.y);
                o3.z = fmaf(c, v3.z, o3.z); o3.w = fmaf(c, v3.w, o3.w);
            }
            if (n0 + n < N) {
                const float4* vf = (const float4*)&qkv[wave][2][f][n][0];
                float4 w0 = vf[0], w1 = vf[1], w2 = vf[2], w3 = vf[3];
                float* dst = out + (((size_t)f * N) + (n0 + n)) * 32;
                *(float4*)(dst +  0) = make_float4(lrelu(o0.x), lrelu(o0.y), lrelu(o0.z), lrelu(o0.w));
                *(float4*)(dst +  4) = make_float4(lrelu(o1.x), lrelu(o1.y), lrelu(o1.z), lrelu(o1.w));
                *(float4*)(dst +  8) = make_float4(lrelu(o2.x), lrelu(o2.y), lrelu(o2.z), lrelu(o2.w));
                *(float4*)(dst + 12) = make_float4(lrelu(o3.x), lrelu(o3.y), lrelu(o3.z), lrelu(o3.w));
                *(float4*)(dst + 16) = make_float4(lrelu(w0.x), lrelu(w0.y), lrelu(w0.z), lrelu(w0.w));
                *(float4*)(dst + 20) = make_float4(lrelu(w1.x), lrelu(w1.y), lrelu(w1.z), lrelu(w1.w));
                *(float4*)(dst + 24) = make_float4(lrelu(w2.x), lrelu(w2.y), lrelu(w2.z), lrelu(w2.w));
                *(float4*)(dst + 28) = make_float4(lrelu(w3.x), lrelu(w3.y), lrelu(w3.z), lrelu(w3.w));
            }
        }
    }
}

extern "C" void kernel_launch(void* const* d_in, const int* in_sizes, int n_in,
                              void* d_out, int out_size, void* d_ws, size_t ws_size,
                              hipStream_t stream) {
    const float* x  = (const float*)d_in[0];
    const float* wq = (const float*)d_in[1];
    const float* wk = (const float*)d_in[2];
    const float* wv = (const float*)d_in[3];
    float* outp = (float*)d_out;

    const int N = in_sizes[0] / (F_FIELDS * 64);       // 500000
    const int ntiles = (N + 15) / 16;                  // 31250
    const int blocks = (ntiles + WPB - 1) / WPB;       // 7813

    autoint_attn_kernel<<<blocks, 32 * WPB, 0, stream>>>(x, wq, wk, wv, outp, N, ntiles);
}
